// JanossyPoolingNonbonded_1408749273398
// MI455X (gfx1250) — matrix operand order repaired and checked by reading, not compiled, not run on hardware
//
#include <hip/hip_runtime.h>

typedef _Float16 v16h __attribute__((ext_vector_type(16)));
typedef _Float16 v8h  __attribute__((ext_vector_type(8)));
typedef _Float16 v4h  __attribute__((ext_vector_type(4)));
typedef float    v8f  __attribute__((ext_vector_type(8)));

#define N_OF 125000
#define N_NB 500000
#define NT_OF ((N_OF + 15) / 16)   // 7813 tiles
#define NT_NB ((N_NB + 15) / 16)   // 31250 tiles
#define NT_TOT (NT_OF + NT_NB)     // 39063
#define WAVES 4
#define XIN_STRIDE  264            // 256 + 8 halfs pad -> rotates LDS banks per row
#define XMID_STRIDE 136            // 128 + 8 halfs pad
#define SLAB_HALFS (16 * XIN_STRIDE + 16 * XMID_STRIDE)  // 6400 halfs per wave
#define LVL_WPACK_HALFS 65536      // W1(32768) + W2(16384) + W3(16384) halfs per level

struct LevelParams {
  const float* b1; const float* b2; const float* b3;
  const float* Ws; const float* We; const float* bs; const float* be;
  const unsigned short* wpack;     // packed f16 fragments: W1 | W2 | W3
  const int* i0; const int* i1;
  int nPairs; int outS; int outE;
};

struct PackArgs { const float* W[6]; };  // [level*3 + layer], level0=onefour

// ---------------------------------------------------------------------------
// Pack f32 row-major W[K][128] into per-lane WMMA B-fragment order (f16).
// Fragment (kt, nt): flat = (kt*8 + nt)*512 + lane*16 + e
//   element e of lane -> K = kt*32 + (lane<16 ? 0 : 8) + (e<8 ? e : 16 + e - 8), N = nt*16 + lane%16
// ---------------------------------------------------------------------------
__global__ void pack_weights_kernel(PackArgs pa, unsigned short* outRaw) {
  int t = blockIdx.x * blockDim.x + threadIdx.x;
  if (t >= 2 * LVL_WPACK_HALFS) return;
  _Float16* out = (_Float16*)outRaw;
  int level = t >> 16;
  int i = t & 0xFFFF;
  int layer, base;
  if (i < 32768)      { layer = 0; base = 0; }
  else if (i < 49152) { layer = 1; base = 32768; }
  else                { layer = 2; base = 49152; }
  const float* W = pa.W[level * 3 + layer];
  int j    = i - base;
  int frag = j >> 9;
  int kt   = frag >> 3;
  int nt   = frag & 7;
  int lane = (j >> 4) & 31;
  int e    = j & 15;
  int K = kt * 32 + ((lane < 16) ? 0 : 8) + ((e < 8) ? e : (16 + e - 8));
  int N = nt * 16 + (lane & 15);
  out[t] = (_Float16)W[K * 128 + N];
}

// ---------------------------------------------------------------------------
// 16 x (KT*32) @ (KT*32) x 128 -> 8 C-tiles of 16x16 f32 via WMMA f16.
// Per k-step: batch-load all 8 B fragments (claused global_load_b128s, one
// wait), then fire 8 back-to-back WMMAs on independent accumulators.
// A fragments from LDS row-major x (stride xstride halfs):
//   lane<16: K base 0 ; lane>=16: K base 8 ; second 8 halfs at +16.
// colAdd rotates k-tiles so ordering B reuses the [h0|h1] gather as [h1|h0].
// ---------------------------------------------------------------------------
__device__ __forceinline__ void mm_tiles(const _Float16* __restrict__ x, int xstride,
                                         int KT, int colAdd,
                                         const _Float16* __restrict__ wp,
                                         int lane, v8f acc[8]) {
#pragma unroll
  for (int nt = 0; nt < 8; ++nt)
#pragma unroll
    for (int i = 0; i < 8; ++i) acc[nt][i] = 0.f;

  const int row  = lane & 15;
  const int ksel = (lane < 16) ? 0 : 8;
  for (int kt = 0; kt < KT; ++kt) {
    const int colbase = ((kt + colAdd) & 7) * 32;
    const _Float16* ap = x + row * xstride + colbase + ksel;
    const _Float16* bp = wp + (size_t)kt * (8 * 512) + lane * 16;

    v16h b[8];
#pragma unroll
    for (int nt = 0; nt < 8; ++nt) b[nt] = *(const v16h*)(bp + nt * 512);

    v8h alo = *(const v8h*)(ap);
    v8h ahi = *(const v8h*)(ap + 16);
    v16h a = __builtin_shufflevector(alo, ahi, 0,1,2,3,4,5,6,7,8,9,10,11,12,13,14,15);

#pragma unroll
    for (int nt = 0; nt < 8; ++nt) {
      acc[nt] = __builtin_amdgcn_wmma_f32_16x16x32_f16(
          false, a, false, b[nt], (short)0, acc[nt], false, false);
    }
  }
}

// C-fragment layout: element r of lane -> row = r + (lane<16?0:8), col = lane%16 (+nt*16)
__device__ __forceinline__ void relu_store(const v8f acc[8], const float* __restrict__ bias,
                                           int lane, _Float16* __restrict__ xout) {
  const int col   = lane & 15;
  const int rbase = (lane < 16) ? 0 : 8;
#pragma unroll
  for (int nt = 0; nt < 8; ++nt) {
    float bv = bias[nt * 16 + col];
#pragma unroll
    for (int r = 0; r < 8; ++r) {
      float v = acc[nt][r] + bv;
      v = v > 0.f ? v : 0.f;
      xout[(rbase + r) * XMID_STRIDE + nt * 16 + col] = (_Float16)v;
    }
  }
}

// Fold layer-3 bias+ReLU AND the two 128->1 head dot products into one
// epilogue: sig/eps accumulate across both Janossy orderings (heads are
// linear, so summing per-ordering head partials == head of summed y).
__device__ __forceinline__ void head_accum(const v8f acc[8], const float* __restrict__ b3,
                                           const float* __restrict__ Ws,
                                           const float* __restrict__ We,
                                           int lane, float sig[8], float eps[8]) {
  const int col = lane & 15;
#pragma unroll
  for (int nt = 0; nt < 8; ++nt) {
    float bv = b3[nt * 16 + col];
    float ws = Ws[nt * 16 + col];
    float we = We[nt * 16 + col];
#pragma unroll
    for (int r = 0; r < 8; ++r) {
      float v = acc[nt][r] + bv;
      v = v > 0.f ? v : 0.f;
      sig[r] += v * ws;
      eps[r] += v * we;
    }
  }
}

// ---------------------------------------------------------------------------
// One wave = one tile of 16 pairs, full 2-ordering MLP + heads.
// ---------------------------------------------------------------------------
__global__ __launch_bounds__(WAVES * 32)
void janossy_mlp_kernel(const float* __restrict__ h, LevelParams L0, LevelParams L1,
                        float* __restrict__ out) {
  const int lane = threadIdx.x & 31;
  const int wv   = threadIdx.x >> 5;
  const int task = blockIdx.x * WAVES + wv;
  if (task >= NT_TOT) return;

  const bool isOF = (task < NT_OF);
  const LevelParams P = isOF ? L0 : L1;
  const int tile = isOF ? task : (task - NT_OF);

  extern __shared__ _Float16 smem[];
  _Float16* xin  = smem + wv * SLAB_HALFS;   // 16 x 256 (+pad) f16: [h0 | h1]
  _Float16* xmid = xin + 16 * XIN_STRIDE;    // 16 x 128 (+pad) f16 intermediate

  const int pairBase = tile * 16;

  // Gather: 16 pairs -> f16 LDS. h (10 MB) is L2-resident, so this hits cache.
  for (int r = 0; r < 16; ++r) {
    int p = pairBase + r;
    if (p >= P.nPairs) p = P.nPairs - 1;
    const float* s0 = h + (size_t)P.i0[p] * 128;
    const float* s1 = h + (size_t)P.i1[p] * 128;
    float4 a = *(const float4*)(s0 + lane * 4);
    float4 b = *(const float4*)(s1 + lane * 4);
    v4h ha = { (_Float16)a.x, (_Float16)a.y, (_Float16)a.z, (_Float16)a.w };
    v4h hb = { (_Float16)b.x, (_Float16)b.y, (_Float16)b.z, (_Float16)b.w };
    *(v4h*)(xin + r * XIN_STRIDE + lane * 4)       = ha;
    *(v4h*)(xin + r * XIN_STRIDE + 128 + lane * 4) = hb;
  }
  // Same-wave LDS ops are in-order; no barrier needed (per-wave private slab).

  const _Float16* wp = (const _Float16*)P.wpack;
  float sig[8], eps[8];
#pragma unroll
  for (int r = 0; r < 8; ++r) { sig[r] = 0.f; eps[r] = 0.f; }

  // Janossy: ordering A = [h0|h1] (colAdd=0), ordering B = [h1|h0] (colAdd=4)
  for (int o = 0; o < 2; ++o) {
    v8f acc[8];
    mm_tiles(xin,  XIN_STRIDE,  8, o * 4, wp,         lane, acc);  // 256->128
    relu_store(acc, P.b1, lane, xmid);
    mm_tiles(xmid, XMID_STRIDE, 4, 0,     wp + 32768, lane, acc);  // 128->128
    relu_store(acc, P.b2, lane, xmid);
    mm_tiles(xmid, XMID_STRIDE, 4, 0,     wp + 49152, lane, acc);  // 128->128
    head_accum(acc, P.b3, P.Ws, P.We, lane, sig, eps);
  }

  // Reduce head partials across the 16 lanes sharing a C-fragment row group.
#pragma unroll
  for (int m = 1; m < 16; m <<= 1) {
#pragma unroll
    for (int r = 0; r < 8; ++r) {
      sig[r] += __shfl_xor(sig[r], m, 32);
      eps[r] += __shfl_xor(eps[r], m, 32);
    }
  }
  if ((lane & 15) == 0) {               // lane 0 -> rows 0..7, lane 16 -> rows 8..15
    const int rbase = (lane >> 4) * 8;
    const float bs0 = P.bs[0], be0 = P.be[0];
    for (int r = 0; r < 8; ++r) {
      int p = pairBase + rbase + r;
      if (p < P.nPairs) {
        out[P.outS + p] = sig[r] + bs0;
        out[P.outE + p] = eps[r] + be0;
      }
    }
  }
}

// ---------------------------------------------------------------------------
extern "C" void kernel_launch(void* const* d_in, const int* in_sizes, int n_in,
                              void* d_out, int out_size, void* d_ws, size_t ws_size,
                              hipStream_t stream) {
  const float* h = (const float*)d_in[0];
  const float *W1[2], *b1[2], *W2[2], *b2[2], *W3[2], *b3[2];
  const float *Ws[2], *bs[2], *We[2], *be[2];
  const int *iof0, *iof1, *inb0, *inb1;

  // Two plausible flattenings of the setup_inputs() pytree:
  //  A) insertion order: h, onefour{W1,b1,W2,b2,W3,b3,Ws,bs,We,be}, nonbonded{...}, idx_of0..idx_nb1
  //  B) sorted keys:     h, idx_nb0, idx_nb1, idx_of0, idx_of1,
  //                      nonbonded{W1,W2,W3,We,Ws,b1,b2,b3,be,bs}, onefour{...}
  const bool sortedLayout = (n_in > 1 && in_sizes[1] == N_NB);
  if (!sortedLayout) {
    for (int L = 0; L < 2; ++L) {            // L=0 onefour, L=1 nonbonded
      int base = 1 + L * 10;
      W1[L] = (const float*)d_in[base + 0]; b1[L] = (const float*)d_in[base + 1];
      W2[L] = (const float*)d_in[base + 2]; b2[L] = (const float*)d_in[base + 3];
      W3[L] = (const float*)d_in[base + 4]; b3[L] = (const float*)d_in[base + 5];
      Ws[L] = (const float*)d_in[base + 6]; bs[L] = (const float*)d_in[base + 7];
      We[L] = (const float*)d_in[base + 8]; be[L] = (const float*)d_in[base + 9];
    }
    iof0 = (const int*)d_in[21]; iof1 = (const int*)d_in[22];
    inb0 = (const int*)d_in[23]; inb1 = (const int*)d_in[24];
  } else {
    inb0 = (const int*)d_in[1]; inb1 = (const int*)d_in[2];
    iof0 = (const int*)d_in[3]; iof1 = (const int*)d_in[4];
    const int bases[2] = { 15, 5 };          // onefour at 15, nonbonded at 5
    for (int L = 0; L < 2; ++L) {
      int base = bases[L];
      W1[L] = (const float*)d_in[base + 0]; W2[L] = (const float*)d_in[base + 1];
      W3[L] = (const float*)d_in[base + 2]; We[L] = (const float*)d_in[base + 3];
      Ws[L] = (const float*)d_in[base + 4]; b1[L] = (const float*)d_in[base + 5];
      b2[L] = (const float*)d_in[base + 6]; b3[L] = (const float*)d_in[base + 7];
      be[L] = (const float*)d_in[base + 8]; bs[L] = (const float*)d_in[base + 9];
    }
  }

  unsigned short* wpack = (unsigned short*)d_ws;   // 256 KB of packed f16 weights
  PackArgs pa;
  for (int L = 0; L < 2; ++L) {
    pa.W[L * 3 + 0] = W1[L]; pa.W[L * 3 + 1] = W2[L]; pa.W[L * 3 + 2] = W3[L];
  }
  pack_weights_kernel<<<(2 * LVL_WPACK_HALFS + 255) / 256, 256, 0, stream>>>(pa, wpack);

  LevelParams L0, L1;
  L0 = { b1[0], b2[0], b3[0], Ws[0], We[0], bs[0], be[0],
         wpack, iof0, iof1, N_OF, 0, N_OF };
  L1 = { b1[1], b2[1], b3[1], Ws[1], We[1], bs[1], be[1],
         wpack + LVL_WPACK_HALFS, inb0, inb1, N_NB, 2 * N_OF, 2 * N_OF + N_NB };

  const int blocks = (NT_TOT + WAVES - 1) / WAVES;
  const size_t shmem = (size_t)WAVES * SLAB_HALFS * sizeof(unsigned short); // 51200 B
  janossy_mlp_kernel<<<blocks, WAVES * 32, shmem, stream>>>(h, L0, L1, (float*)d_out);

  (void)in_sizes; (void)n_in; (void)out_size; (void)ws_size;
}
